// InLineAttention_30717606101231
// MI455X (gfx1250) — matrix operand order repaired and checked by reading, not compiled, not run on hardware
//
#include <hip/hip_runtime.h>
#include <hip/hip_bf16.h>
#include <math.h>

// ---------------------------------------------------------------------------
// InLineAttention for MI455X (gfx1250), wave32 + WMMA bf16.
// B=8, T=24, N=512, C=512, H=8, D=64  ->  BT=192, rows R = BT*N = 98304.
// All matmuls use V_WMMA_F32_16X16X32_BF16 with fp32 accumulation.
// GEMM A-tiles / einsum operands staged in LDS (padded rows, bank-conflict-free);
// B fragments double-buffered in registers so WMMAs overlap global loads.
// ---------------------------------------------------------------------------

typedef __attribute__((ext_vector_type(16))) __bf16 v16bf;
typedef __attribute__((ext_vector_type(8)))  __bf16 v8bf;
typedef __attribute__((ext_vector_type(4)))  __bf16 v4bf;
typedef __attribute__((ext_vector_type(8)))  float  v8f;

#define BT_   192
#define NN_   512
#define CC_   512
#define HH_   8
#define DD_   64
#define RR_   (BT_ * NN_)        // 98304
#define LPAD_ 8                  // LDS row pad (bf16 elems) for bank spread

// A-fragment (16x32 bf16, row-major source, lane l = row m0+l):
// lane holds K = {hi*16 .. hi*16+15} (contiguous, matches B fragment)
__device__ __forceinline__ v16bf load_a_frag(const __bf16* rowk0, int hi) {
  const __bf16* p = rowk0 + hi * 16;
  v8bf lo = *(const v8bf*)(p);
  v8bf hh = *(const v8bf*)(p + 8);
  v16bf a;
#pragma unroll
  for (int t = 0; t < 8; ++t) { a[t] = lo[t]; a[t + 8] = hh[t]; }
  return a;
}

// B-fragment (32x16 bf16) from K-major (transposed) matrix Bt[n][k]:
// lane l = column n0+l, holds K = {hi*16 .. hi*16+15} packed in pairs.
__device__ __forceinline__ v16bf load_b_frag(const __bf16* coln_k0, int hi) {
  return *(const v16bf*)(coln_k0 + hi * 16);
}

__device__ __forceinline__ v8f wmma_bf16(v16bf a, v16bf b, v8f c) {
  return __builtin_amdgcn_wmma_f32_16x16x32_bf16(false, a, false, b, (short)0, c,
                                                 false, false);
}

// ---------------------------------------------------------------------------
// transpose+convert the four 512x512 fp32 weight matrices to bf16 K-major
// ---------------------------------------------------------------------------
__global__ __launch_bounds__(256) void wt_prep_kernel(
    const float* __restrict__ Wq, const float* __restrict__ Wk,
    const float* __restrict__ Wv, const float* __restrict__ Wo,
    __bf16* __restrict__ WqT, __bf16* __restrict__ WkT,
    __bf16* __restrict__ WvT, __bf16* __restrict__ WoT) {
  int idx = blockIdx.x * 256 + threadIdx.x;          // < 4 * 512 * 512
  int w = idx >> 18;
  int within = idx & 262143;
  int i = within & 511;                              // K index (fastest -> coalesced write)
  int o = within >> 9;                               // output column
  const float* S = (w == 0) ? Wq : (w == 1) ? Wk : (w == 2) ? Wv : Wo;
  __bf16* Dst = (w == 0) ? WqT : (w == 1) ? WkT : (w == 2) ? WvT : WoT;
  Dst[within] = (__bf16)S[(size_t)i * 512 + o];      // WT[o][i] = W[i][o]
}

// fp32 -> bf16, 4 elements per thread (float4 in, 8B out)
__global__ __launch_bounds__(256) void cvt_bf16_kernel(const float4* __restrict__ x,
                                                       v4bf* __restrict__ xb,
                                                       long long n4) {
  long long i = (long long)blockIdx.x * 256 + threadIdx.x;
  if (i >= n4) return;
  float4 f = x[i];
  v4bf o;
  o[0] = (__bf16)f.x; o[1] = (__bf16)f.y; o[2] = (__bf16)f.z; o[3] = (__bf16)f.w;
  xb[i] = o;
}

// xm[bt][c] = mean over n of x[bt][n][c]
__global__ __launch_bounds__(256) void xm_kernel(const float* __restrict__ x,
                                                 float* __restrict__ xm) {
  int idx = blockIdx.x * 256 + threadIdx.x;          // < BT*C
  if (idx >= BT_ * CC_) return;
  int c = idx & 511, bt = idx >> 9;
  const float* xp = x + (size_t)bt * NN_ * CC_ + c;
  float s = 0.f;
  for (int n = 0; n < NN_; ++n) s += xp[(size_t)n * CC_];
  xm[idx] = s * (1.0f / NN_);
}

// ---------------------------------------------------------------------------
// Main WMMA GEMM: Y(R x 512) = A(R x 512, bf16 row-major) @ B + bias
// Block = 256 thr = 8 waves; all waves share one 16-row A tile (LDS-staged),
// wave w owns 64-column strip c0 = 64*w. B frags double-buffered in registers.
// ---------------------------------------------------------------------------
__global__ __launch_bounds__(256) void wmma_gemm512_kernel(
    const __bf16* __restrict__ A, const __bf16* __restrict__ Bt,
    const float* __restrict__ bias, __bf16* __restrict__ outN,
    __bf16* __restrict__ outT, float* __restrict__ outF) {
  __shared__ __attribute__((aligned(32))) __bf16 As[16 * (512 + LPAD_)];
  const int lane = threadIdx.x & 31;
  const int wave = threadIdx.x >> 5;
  const int m0 = blockIdx.x * 16;                    // 6144 blocks
  const int c0 = wave * 64;
  const int l = lane & 15;
  const int hi = lane >> 4;

  // cooperative stage of A tile: 16 rows x 512 bf16 (64 uint4 per row)
  {
    const uint4* src = (const uint4*)(A + (size_t)m0 * 512);
#pragma unroll
    for (int ch = threadIdx.x; ch < 1024; ch += 256) {
      int r = ch >> 6, j = ch & 63;
      *((uint4*)(As + r * (512 + LPAD_)) + j) = src[(size_t)r * 64 + j];
    }
  }
  __syncthreads();

  v8f acc[4] = {};
  v16bf bc[4], bn[4];
  const __bf16* arow = As + (size_t)l * (512 + LPAD_);
  const __bf16* brow0 = Bt + (size_t)(c0 + l) * 512;
#pragma unroll
  for (int t = 0; t < 4; ++t) bc[t] = load_b_frag(brow0 + (size_t)t * 16 * 512, hi);

#pragma unroll
  for (int k0 = 0; k0 < 512; k0 += 32) {
    if (k0 < 480) {
#pragma unroll
      for (int t = 0; t < 4; ++t)
        bn[t] = load_b_frag(brow0 + (size_t)t * 16 * 512 + k0 + 32, hi);
    }
    v16bf a = load_a_frag(arow + k0, hi);
#pragma unroll
    for (int t = 0; t < 4; ++t) acc[t] = wmma_bf16(a, bc[t], acc[t]);
    if (k0 < 480) {
#pragma unroll
      for (int t = 0; t < 4; ++t) bc[t] = bn[t];
    }
  }

  const int mbase = m0 + hi * 8;                     // D-matrix row mapping (ISA 7.12.2)
#pragma unroll
  for (int t = 0; t < 4; ++t) {
    const int c = c0 + t * 16 + l;
    const float bv = bias[c];
#pragma unroll
    for (int r = 0; r < 8; ++r) {
      const int m = mbase + r;
      float val = acc[t][r] + bv;
      if (outN) outN[(size_t)m * 512 + c] = (__bf16)val;
      if (outT) {
        int bt = m >> 9, n = m & 511, h = c >> 6, d = c & 63;
        outT[(((size_t)(bt * HH_ + h) * DD_ + d) << 9) + n] = (__bf16)val;
      }
      if (outF) outF[(size_t)m * 512 + c] = val;
    }
  }
}

// ---------------------------------------------------------------------------
// kv einsum: kv[bt,h,d,e] = s^2 * sum_n k[n,d] * v[n,e]    (K = N = 512)
// Block = (bt,h); both 64x512 operand strips staged in LDS (shared by 4 waves).
// Writes kvT[bt][h][e][d] bf16 (transposed for the attn GEMM's B side).
// ---------------------------------------------------------------------------
__global__ __launch_bounds__(128) void wmma_kv_kernel(const __bf16* __restrict__ kT,
                                                      const __bf16* __restrict__ vT,
                                                      __bf16* __restrict__ kvT,
                                                      float s2) {
  __shared__ __attribute__((aligned(32))) __bf16 Ks[64 * (512 + LPAD_)];
  __shared__ __attribute__((aligned(32))) __bf16 Vs[64 * (512 + LPAD_)];
  const int bh = blockIdx.x;                         // bt*H + h
  const int wave = threadIdx.x >> 5;
  const int lane = threadIdx.x & 31;
  const int l = lane & 15, hi = lane >> 4;
  const int d0 = wave * 16;

  {
    const uint4* ksrc = (const uint4*)(kT + (size_t)bh * DD_ * NN_);
    const uint4* vsrc = (const uint4*)(vT + (size_t)bh * DD_ * NN_);
#pragma unroll
    for (int ch = threadIdx.x; ch < 4096; ch += 128) {     // 64 rows x 64 uint4
      int r = ch >> 6, j = ch & 63;
      *((uint4*)(Ks + r * (512 + LPAD_)) + j) = ksrc[(size_t)r * 64 + j];
      *((uint4*)(Vs + r * (512 + LPAD_)) + j) = vsrc[(size_t)r * 64 + j];
    }
  }
  __syncthreads();

  v8f acc[4] = {};
  const __bf16* arow = Ks + (size_t)(d0 + l) * (512 + LPAD_);
  const __bf16* brow0 = Vs + (size_t)l * (512 + LPAD_);
#pragma unroll
  for (int n0 = 0; n0 < NN_; n0 += 32) {
    v16bf a = load_a_frag(arow + n0, hi);
#pragma unroll
    for (int t = 0; t < 4; ++t) {
      v16bf b = load_b_frag(brow0 + (size_t)t * 16 * (512 + LPAD_) + n0, hi);
      acc[t] = wmma_bf16(a, b, acc[t]);
    }
  }

  __bf16* kvb = kvT + (size_t)bh * DD_ * DD_;
#pragma unroll
  for (int t = 0; t < 4; ++t) {
    const int e = t * 16 + l;
#pragma unroll
    for (int r = 0; r < 8; ++r) {
      const int d = d0 + hi * 8 + r;
      kvb[(size_t)e * DD_ + d] = (__bf16)(acc[t][r] * s2);   // store transposed
    }
  }
}

// means over n from the K-major per-head layouts
__global__ __launch_bounds__(256) void means_kernel(const __bf16* __restrict__ kT,
                                                    const __bf16* __restrict__ vT,
                                                    float* __restrict__ kmean,
                                                    float* __restrict__ vmean) {
  int idx = blockIdx.x * 256 + threadIdx.x;          // < 2 * BT*H*D
  if (idx >= 2 * BT_ * HH_ * DD_) return;
  int which = (idx >= BT_ * HH_ * DD_);
  int row = idx - which * (BT_ * HH_ * DD_);         // bh*64 + d
  const __bf16* src = (which ? vT : kT) + (size_t)row * NN_;
  float s = 0.f;
  for (int n = 0; n < NN_; ++n) s += (float)src[n];
  float m = s * (1.0f / NN_);
  if (which) vmean[row] = m; else kmean[row] = m;
}

// grouped 1x1-conv net: h1 = gelu(xm_h @ W1_h^T + b1); h2 = h1 @ W2_h^T + b2 -> w3
__global__ __launch_bounds__(256) void w3_kernel(const float* __restrict__ xm,
                                                 const float* __restrict__ W1,
                                                 const float* __restrict__ b1,
                                                 const float* __restrict__ W2,
                                                 const float* __restrict__ b2,
                                                 float* __restrict__ w3) {
  __shared__ float xs[CC_];
  __shared__ float h1[CC_];
  const int bt = blockIdx.x, t = threadIdx.x;
  for (int i = t; i < CC_; i += 256) xs[i] = xm[(size_t)bt * CC_ + i];
  __syncthreads();
  for (int oc = t; oc < CC_; oc += 256) {            // oc = h*64 + o
    int h = oc >> 6;
    float s = b1[oc];
    const float* wr = W1 + (size_t)oc * DD_;
    const float* xh = xs + h * DD_;
    for (int i = 0; i < DD_; ++i) s += xh[i] * wr[i];
    h1[oc] = 0.5f * s * (1.0f + erff(s * 0.7071067811865475f));  // exact gelu
  }
  __syncthreads();
  for (int oc = t; oc < 3 * CC_; oc += 256) {        // oc = h*192 + o
    int h = oc / 192;
    float s = b2[oc];
    const float* wr = W2 + (size_t)oc * DD_;
    const float* hh = h1 + h * DD_;
    for (int i = 0; i < DD_; ++i) s += hh[i] * wr[i];
    w3[(size_t)bt * (3 * CC_) + oc] = s;             // flat (H,3D) == (C,3)
  }
}

// qkm[bt,h,n] = q[bt,n,h,:] . k_mean[bt,h,:]
__global__ __launch_bounds__(256) void qkm_kernel(const __bf16* __restrict__ qbf,
                                                  const float* __restrict__ kmean,
                                                  float* __restrict__ qkm) {
  int idx = blockIdx.x * 256 + threadIdx.x;          // < BT*H*N
  if (idx >= BT_ * HH_ * NN_) return;
  int n = idx & 511, bh = idx >> 9;
  int h = bh & 7, bt = bh >> 3;
  const __bf16* qp = qbf + ((size_t)(bt * NN_ + n)) * CC_ + h * DD_;
  const float* km = kmean + (size_t)bh * DD_;
  float s = 0.f;
  for (int d = 0; d < DD_; ++d) s += (float)qp[d] * km[d];
  qkm[idx] = s;
}

// extra[bt,n,c] = (1 - qkm*scale)*v_mean + w3.(depthwise length-3 conv of v over n)
__global__ __launch_bounds__(256) void extra_kernel(const float* __restrict__ qkm,
                                                    const float* __restrict__ vmean,
                                                    const float* __restrict__ w3,
                                                    const __bf16* __restrict__ vbf,
                                                    __bf16* __restrict__ extra,
                                                    float scale) {
  long long idx = (long long)blockIdx.x * 256 + threadIdx.x;  // < BT*N*C
  if (idx >= (long long)BT_ * NN_ * CC_) return;
  int c = (int)(idx & 511);
  int n = (int)((idx >> 9) & 511);
  int bt = (int)(idx >> 18);
  int h = c >> 6, e = c & 63;
  int bh = bt * HH_ + h;
  float qv = qkm[(size_t)bh * NN_ + n];
  float vm = vmean[(size_t)bh * DD_ + e];
  const float* w = w3 + (size_t)bt * (3 * CC_) + (size_t)c * 3;
  const __bf16* vp = vbf + ((size_t)(bt * NN_ + n)) * CC_ + c;
  float vprev = (n > 0)        ? (float)vp[-(int)CC_] : 0.f;
  float vcur  = (float)vp[0];
  float vnext = (n < NN_ - 1)  ? (float)vp[CC_]       : 0.f;
  extra[idx] = (__bf16)((1.f - qv * scale) * vm + w[0] * vprev + w[1] * vcur + w[2] * vnext);
}

// attn GEMM: per (bt,h):  xa = q (N x D) @ kv (D x D) + extra ; K = 64.
// kv tile (8 KB) staged in LDS, shared by the block's 4 waves.
__global__ __launch_bounds__(128) void wmma_attn_kernel(const __bf16* __restrict__ qbf,
                                                        const __bf16* __restrict__ kvT,
                                                        const __bf16* __restrict__ extra,
                                                        __bf16* __restrict__ xa) {
  __shared__ __attribute__((aligned(32))) __bf16 KVs[64 * (64 + LPAD_)];
  const int chunk = blockIdx.x & 7;                  // 8 chunks of 64 rows
  const int bh = blockIdx.x >> 3;                    // bt*H + h
  const int bt = bh >> 3, h = bh & 7;
  const int wave = threadIdx.x >> 5;
  const int lane = threadIdx.x & 31;
  const int l = lane & 15, hi = lane >> 4;
  const int n0r = (chunk * 4 + wave) * 16;           // row tile base

  {
    const uint4* src = (const uint4*)(kvT + (size_t)bh * DD_ * DD_);
#pragma unroll
    for (int ch = threadIdx.x; ch < 512; ch += 128) {      // 64 rows x 8 uint4
      int r = ch >> 3, j = ch & 7;
      *((uint4*)(KVs + r * (64 + LPAD_)) + j) = src[(size_t)r * 8 + j];
    }
  }
  __syncthreads();

  const __bf16* arow = qbf + ((size_t)(bt * NN_ + n0r + l)) * CC_ + h * DD_;
  const __bf16* brow0 = KVs + (size_t)l * (64 + LPAD_);

  v8f acc[4] = {};
#pragma unroll
  for (int k0 = 0; k0 < DD_; k0 += 32) {
    v16bf a = load_a_frag(arow + k0, hi);
#pragma unroll
    for (int t = 0; t < 4; ++t) {
      v16bf b = load_b_frag(brow0 + (size_t)t * 16 * (64 + LPAD_) + k0, hi);
      acc[t] = wmma_bf16(a, b, acc[t]);
    }
  }
#pragma unroll
  for (int t = 0; t < 4; ++t) {
    const int c = h * DD_ + t * 16 + l;
#pragma unroll
    for (int r = 0; r < 8; ++r) {
      const int n = n0r + hi * 8 + r;
      size_t o = ((size_t)(bt * NN_ + n)) * CC_ + c;
      xa[o] = (__bf16)(acc[t][r] + (float)extra[o]);
    }
  }
}

// ---------------------------------------------------------------------------
extern "C" void kernel_launch(void* const* d_in, const int* in_sizes, int n_in,
                              void* d_out, int out_size, void* d_ws, size_t ws_size,
                              hipStream_t stream) {
  const float* x  = (const float*)d_in[0];
  const float* Wq = (const float*)d_in[1];
  const float* bq = (const float*)d_in[2];
  const float* Wk = (const float*)d_in[3];
  const float* bk = (const float*)d_in[4];
  const float* Wv = (const float*)d_in[5];
  const float* bv = (const float*)d_in[6];
  const float* W1 = (const float*)d_in[7];
  const float* b1 = (const float*)d_in[8];
  const float* W2 = (const float*)d_in[9];
  const float* b2 = (const float*)d_in[10];
  const float* Wo = (const float*)d_in[11];
  const float* bo = (const float*)d_in[12];
  float* out = (float*)d_out;

  // --- workspace carve-up (256B aligned slices) ---
  size_t off = 0;
  char* base = (char*)d_ws;
  auto carve = [&](size_t bytes) -> void* {
    void* p = base + off;
    off += (bytes + 255) & ~(size_t)255;
    return p;
  };
  const size_t bigBF = (size_t)RR_ * CC_ * sizeof(__bf16);     // 100.66 MB
  __bf16* WqT = (__bf16*)carve((size_t)CC_ * CC_ * 2);
  __bf16* WkT = (__bf16*)carve((size_t)CC_ * CC_ * 2);
  __bf16* WvT = (__bf16*)carve((size_t)CC_ * CC_ * 2);
  __bf16* WoT = (__bf16*)carve((size_t)CC_ * CC_ * 2);
  __bf16* xbf = (__bf16*)carve(bigBF);                         // reused as xa later
  __bf16* qbf = (__bf16*)carve(bigBF);
  __bf16* vbf = (__bf16*)carve(bigBF);
  __bf16* kT  = (__bf16*)carve(bigBF);
  __bf16* vT  = (__bf16*)carve(bigBF);
  __bf16* ext = (__bf16*)carve(bigBF);
  __bf16* kvT = (__bf16*)carve((size_t)BT_ * HH_ * DD_ * DD_ * 2);
  float* xm    = (float*)carve((size_t)BT_ * CC_ * 4);
  float* kmean = (float*)carve((size_t)BT_ * HH_ * DD_ * 4);
  float* vmean = (float*)carve((size_t)BT_ * HH_ * DD_ * 4);
  float* qkm   = (float*)carve((size_t)BT_ * HH_ * NN_ * 4);
  float* w3    = (float*)carve((size_t)BT_ * 3 * CC_ * 4);
  __bf16* xa = xbf;  // xbf no longer needed once attn runs

  const float scale = 0.125f;                 // D^-1/2
  const float s2 = scale / (float)NN_;        // s^2 applied in kv einsum

  // 1) weights -> bf16 K-major
  wt_prep_kernel<<<4096, 256, 0, stream>>>(Wq, Wk, Wv, Wo, WqT, WkT, WvT, WoT);
  // 2) x -> bf16 (vectorized), node-mean
  cvt_bf16_kernel<<<49152, 256, 0, stream>>>((const float4*)x, (v4bf*)xbf,
                                             (long long)RR_ * CC_ / 4);
  xm_kernel<<<384, 256, 0, stream>>>(x, xm);
  // 3) q / k / v projections (WMMA)
  wmma_gemm512_kernel<<<6144, 256, 0, stream>>>(xbf, WqT, bq, qbf, nullptr, nullptr);
  wmma_gemm512_kernel<<<6144, 256, 0, stream>>>(xbf, WkT, bk, nullptr, kT, nullptr);
  wmma_gemm512_kernel<<<6144, 256, 0, stream>>>(xbf, WvT, bv, vbf, vT, nullptr);
  // 4) per-head means
  means_kernel<<<768, 256, 0, stream>>>(kT, vT, kmean, vmean);
  // 5) kv = s^2 * k^T v  (WMMA, LDS-staged)
  wmma_kv_kernel<<<BT_ * HH_, 128, 0, stream>>>(kT, vT, kvT, s2);
  // 6) dynamic depthwise kernels
  w3_kernel<<<BT_, 256, 0, stream>>>(xm, W1, b1, W2, b2, w3);
  // 7) q . k_mean
  qkm_kernel<<<3072, 256, 0, stream>>>(qbf, kmean, qkm);
  // 8) epilogue buffer: mean-correction + depthwise conv residual
  extra_kernel<<<196608, 256, 0, stream>>>(qkm, vmean, w3, vbf, ext, scale);
  // 9) attn = q @ kv + extra  (WMMA, LDS-staged kv)
  wmma_attn_kernel<<<BT_ * HH_ * 8, 128, 0, stream>>>(qbf, kvT, ext, xa);
  // 10) out = xa @ Wo + bo  (WMMA, fp32 store)
  wmma_gemm512_kernel<<<6144, 256, 0, stream>>>(xa, WoT, bo, nullptr, nullptr, out);

  (void)in_sizes; (void)n_in; (void)out_size; (void)ws_size;
}
